// DGNBlock_R_84799834293153
// MI455X (gfx1250) — matrix operand order repaired and checked by reading, not compiled, not run on hardware
//
#include <hip/hip_runtime.h>
#include <hip/hip_bf16.h>

// ---------------- problem constants ----------------
#define N_   32
#define C_   64
#define T_   300
#define V_   25
#define TV   7500      // T_*V_
#define CT   19200     // C_*T_
#define K3C  192       // 3*C_
#define TVP  7504      // TV padded to multiple of 16
#define EPSF 1e-5f

typedef __attribute__((ext_vector_type(16))) __bf16 v16bf;
typedef __attribute__((ext_vector_type(8)))  float  v8f;

// float -> bf16 (round to nearest even), stored as ushort
__device__ inline unsigned short f2bf(float f) {
    unsigned u = __float_as_uint(f);
    u += 0x7FFFu + ((u >> 16) & 1u);
    return (unsigned short)(u >> 16);
}

__device__ inline v16bf load_ab(const unsigned short* p0, const unsigned short* p1) {
    union { uint4 u[2]; v16bf v; } x;
    x.u[0] = *(const uint4*)p0;
    x.u[1] = *(const uint4*)p1;
    return x.v;
}

// ---------------- K1: CeN channel+time reduction -> x[N,V] ----------------
__global__ void cen_reduce(const float* __restrict__ fv, const float* __restrict__ fe,
                           const float* __restrict__ wc, const float* __restrict__ bc,
                           const float* __restrict__ wt, const float* __restrict__ bt,
                           float* __restrict__ xout) {
    int n = blockIdx.x;
    int lane = threadIdx.x & 31;   // v index
    int grp  = threadIdx.x >> 5;   // 0..7
    const float* fvn = fv + (size_t)n * C_ * T_ * V_;
    const float* fen = fe + (size_t)n * C_ * T_ * V_;
    float acc = 0.f;
    if (lane < V_) {
        for (int t = grp; t < T_; t += 8) {
            float inner = bc[0];
            for (int c = 0; c < C_; ++c) {
                inner += fvn[(c * T_ + t) * V_ + lane] * wc[c];
                inner += fen[(c * T_ + t) * V_ + lane] * wc[C_ + c];
            }
            acc += wt[t] * inner;
        }
    }
    __shared__ float red[8][32];
    red[grp][lane] = acc;
    __syncthreads();
    if (grp == 0 && lane < V_) {
        float s = bt[0];
        for (int g = 0; g < 8; ++g) s += red[g][lane];
        xout[n * V_ + lane] = s;
    }
}

// ---------------- K2a: BN over batch + A = wv @ xhat + bv ----------------
__global__ void cen_bn_A(const float* __restrict__ xin,
                         const float* __restrict__ bn_g, const float* __restrict__ bn_b,
                         const float* __restrict__ wv, const float* __restrict__ bv,
                         float* __restrict__ Aout) {
    __shared__ float xh[N_][V_];
    int tid = threadIdx.x;
    if (tid < V_) {
        float mu = 0.f;
        for (int n = 0; n < N_; ++n) mu += xin[n * V_ + tid];
        mu /= (float)N_;
        float m2 = 0.f;
        for (int n = 0; n < N_; ++n) { float d = xin[n * V_ + tid] - mu; m2 += d * d; }
        m2 /= (float)N_;
        float inv = rsqrtf(m2 + EPSF);
        for (int n = 0; n < N_; ++n)
            xh[n][tid] = (xin[n * V_ + tid] - mu) * inv * bn_g[tid] + bn_b[tid];
    }
    __syncthreads();
    for (int idx = tid; idx < N_ * 625; idx += blockDim.x) {
        int n = idx / 625, o = idx % 625;
        float s = bv[o];
        for (int v = 0; v < V_; ++v) s += wv[o * V_ + v] * xh[n][v];
        Aout[idx] = s;
    }
}

// ---------------- K2b: normalize A, mean, sM/tM, bf16 operand prep ----------------
__global__ void cen_finish(const float* __restrict__ Aout,
                           const float* __restrict__ srcM, const float* __restrict__ tgtM,
                           const float* __restrict__ alpha, const float* __restrict__ alpha1,
                           const float* __restrict__ He_w, const float* __restrict__ Hv_w,
                           float* __restrict__ sMf, float* __restrict__ tMf,
                           unsigned short* __restrict__ sMB,  unsigned short* __restrict__ tMB,
                           unsigned short* __restrict__ sMBT, unsigned short* __restrict__ tMBT,
                           unsigned short* __restrict__ HeWb, unsigned short* __restrict__ HvWb) {
    int tid = threadIdx.x;
    if (tid < 625) {
        int r = tid / V_, cc = tid % V_;
        float m = 0.f;
        for (int n = 0; n < N_; ++n) {
            float ss = 0.f;
            for (int rr = 0; rr < V_; ++rr) {
                float a = Aout[n * 625 + rr * V_ + cc];
                ss += a * a;
            }
            m += Aout[n * 625 + r * V_ + cc] * rsqrtf(ss);
        }
        m /= (float)N_;
        sMf[tid] = srcM[tid] + alpha[0]  * m;
        tMf[tid] = tgtM[tid] + alpha1[0] * m;
    }
    __syncthreads();
    // padded 32x32 WMMA B operands, K-contiguous per column: arr[col*32 + k]
    if (tid < 1024) {
        int col = tid >> 5, k = tid & 31;
        bool ok = (col < V_) && (k < V_);
        // B  : k=v, col=e  -> sM[v,e]          (edge: out[.,e] = sum_v in[.,v]*sM[v,e])
        sMB[tid]  = f2bf(ok ? sMf[k * V_ + col] : 0.f);
        tMB[tid]  = f2bf(ok ? tMf[k * V_ + col] : 0.f);
        // BT : k=e, col=v  -> sM[v,e]          (node: out[.,v] = sum_e in[.,e]*sM[v,e])
        sMBT[tid] = f2bf(ok ? sMf[col * V_ + k] : 0.f);
        tMBT[tid] = f2bf(ok ? tMf[col * V_ + k] : 0.f);
    }
    for (int i = tid; i < C_ * K3C; i += blockDim.x) {
        HeWb[i] = f2bf(He_w[i]);
        HvWb[i] = f2bf(Hv_w[i]);
    }
}

// ---------------- K3: fp32 [n][64][TV] -> bf16 stack columns rows 0..63 ----------------
__global__ void stack_cols(const float* __restrict__ src, unsigned short* __restrict__ STACK) {
    int n = blockIdx.y;
    int col0 = blockIdx.x * 64;
    __shared__ unsigned short tile[64][72];   // [col][c], padded
    const float* s = src + (size_t)n * C_ * TV;
    for (int i = threadIdx.x; i < 64 * 64; i += 256) {
        int c = i >> 6, j = i & 63;
        int col = col0 + j;
        float v = (col < TV) ? s[(size_t)c * TV + col] : 0.f;
        tile[j][c] = f2bf(v);
    }
    __syncthreads();
    unsigned* dst = (unsigned*)STACK;
    for (int p = 0; p < 8; ++p) {
        int j = p * 8 + (threadIdx.x >> 5);
        int u = threadIdx.x & 31;
        int col = col0 + j;
        if (col < TVP) {
            const unsigned* tu = (const unsigned*)&tile[j][0];
            dst[(size_t)(n * TVP + col) * 96 + u] = tu[u];
        }
    }
}

// ---------------- K4: fp32 [rows][25] -> bf16 row-major A buffer [rows][32] padded ----------------
__global__ void to_rowA(const float* __restrict__ src, unsigned short* __restrict__ RA) {
    size_t row = (size_t)blockIdx.x * 8 + (threadIdx.x >> 5);
    int v = threadIdx.x & 31;
    float val = (v < V_) ? src[row * V_ + v] : 0.f;
    RA[row * 32 + v] = f2bf(val);
}

// ---------------- K5: graph matmul via WMMA (K=32 covers V=25 padded) ----------------
__global__ void graph_wmma(const unsigned short* __restrict__ RA,
                           const unsigned short* __restrict__ B1,
                           const unsigned short* __restrict__ B2,
                           unsigned short* __restrict__ STACK) {
    int wave = threadIdx.x >> 5;
    int lane = threadIdx.x & 31;
    int n = blockIdx.y;
    int rtile = blockIdx.x * 8 + wave;            // 0..1199
    const unsigned short* ra = RA + (size_t)n * CT * 32;
    int arow = rtile * 16 + (lane & 15);
    int kb  = (lane < 16) ? 0 : 8;
    int kb2 = (lane < 16) ? 0 : 16;
    v16bf a = load_ab(ra + (size_t)arow * 32 + kb, ra + (size_t)arow * 32 + 16 + kb);
    unsigned short* outn = STACK + (size_t)n * TVP * 192;
    int hi8 = (lane < 16) ? 0 : 8;
    for (int m = 0; m < 2; ++m) {
        const unsigned short* B = m ? B2 : B1;
        int kbase = 64 + m * 64;
        for (int ct2 = 0; ct2 < 2; ++ct2) {
            int e = ct2 * 16 + (lane & 15);
            const unsigned short* bp = B + e * 32 + kb;
            v16bf b = load_ab(bp, bp + 16);
            v8f c = {};
            c = __builtin_amdgcn_wmma_f32_16x16x32_bf16(false, a, false, b, (short)0, c,
                                                        false, false);
            if (e < V_) {
                for (int r = 0; r < 8; ++r) {
                    int ct = rtile * 16 + r + hi8;
                    int cch = ct / T_;
                    int t = ct - cch * T_;
                    outn[(size_t)(t * V_ + e) * 192 + kbase + cch] = f2bf(c[r]);
                }
            }
        }
    }
}

// ---------------- K6: channel GEMM via WMMA, M=64 K=192 N=TVP ----------------
__global__ void gemm_wmma(const unsigned short* __restrict__ Aw,
                          const unsigned short* __restrict__ STACK,
                          const float* __restrict__ bias,
                          float* __restrict__ PRE) {
    int wave = threadIdx.x >> 5;
    int lane = threadIdx.x & 31;
    int n = blockIdx.y;
    int mtile = wave & 3;
    int ntile = blockIdx.x * 2 + (wave >> 2);
    if (ntile * 16 >= TVP) return;                // wave-uniform exit
    int arow = mtile * 16 + (lane & 15);
    int kb  = (lane < 16) ? 0 : 8;
    int kb2 = (lane < 16) ? 0 : 16;
    const unsigned short* bcol =
        STACK + (size_t)n * TVP * 192 + (size_t)(ntile * 16 + (lane & 15)) * 192;
    v8f c = {};
#pragma unroll
    for (int s = 0; s < 6; ++s) {
        const unsigned short* ap = Aw + (size_t)arow * 192 + s * 32;
        v16bf a = load_ab(ap + kb, ap + 16 + kb);
        const unsigned short* bp = bcol + s * 32 + kb;
        v16bf b = load_ab(bp, bp + 16);
        c = __builtin_amdgcn_wmma_f32_16x16x32_bf16(false, a, false, b, (short)0, c,
                                                    false, false);
    }
    int ncol = ntile * 16 + (lane & 15);
    float* pn = PRE + (size_t)n * C_ * TV;
    int hi8 = (lane < 16) ? 0 : 8;
    for (int r = 0; r < 8; ++r) {
        int o = mtile * 16 + r + hi8;
        if (ncol < TV) pn[(size_t)o * TV + ncol] = c[r] + bias[o];
    }
}

// ---------------- K7: BN stats, stage 1 (deterministic, no atomics) ----------------
__global__ void bn_stats_partial(const float* __restrict__ PRE, float* __restrict__ partial) {
    int o = blockIdx.x;
    int n = blockIdx.y;
    const float* p = PRE + ((size_t)n * C_ + o) * TV;
    float s = 0.f, s2 = 0.f;
    for (int i = threadIdx.x; i < TV; i += 256) { float v = p[i]; s += v; s2 += v * v; }
    __shared__ float rs[256], rq[256];
    rs[threadIdx.x] = s; rq[threadIdx.x] = s2;
    __syncthreads();
    for (int st = 128; st > 0; st >>= 1) {
        if (threadIdx.x < st) { rs[threadIdx.x] += rs[threadIdx.x + st];
                                rq[threadIdx.x] += rq[threadIdx.x + st]; }
        __syncthreads();
    }
    if (threadIdx.x == 0) {
        partial[(o * N_ + n) * 2]     = rs[0];
        partial[(o * N_ + n) * 2 + 1] = rq[0];
    }
}

// ---------------- K8: BN stats, stage 2 ----------------
__global__ void bn_stats_combine(const float* __restrict__ partial, float* __restrict__ stats) {
    int o = threadIdx.x;
    if (o >= C_) return;
    float s = 0.f, s2 = 0.f;
    for (int n = 0; n < N_; ++n) {
        s  += partial[(o * N_ + n) * 2];
        s2 += partial[(o * N_ + n) * 2 + 1];
    }
    float inv = 1.f / (float)(N_ * TV);
    float mu  = s * inv;
    float var = s2 * inv - mu * mu;
    stats[o]      = mu;
    stats[C_ + o] = rsqrtf(var + EPSF);
}

// ---------------- K9: BN + ReLU + residual (+ optional bf16 A-layout emit) ----------------
__global__ void bn_finalize(const float* __restrict__ PRE, const float* __restrict__ stats,
                            const float* __restrict__ g, const float* __restrict__ bb,
                            const float* __restrict__ addend, float* __restrict__ outp,
                            unsigned short* __restrict__ RA, int writeRA) {
    size_t idx = (size_t)blockIdx.x * 256 + threadIdx.x;
    if (idx >= (size_t)N_ * C_ * TV) return;
    size_t ctv = idx % ((size_t)C_ * TV);
    int o  = (int)(ctv / TV);
    int tv = (int)(ctv % TV);
    float y = (PRE[idx] - stats[o]) * stats[C_ + o] * g[o] + bb[o];
    y = y > 0.f ? y : 0.f;
    outp[idx] = y + addend[idx];
    if (writeRA) {
        int n = (int)(idx / ((size_t)C_ * TV));
        int t = tv / V_, v = tv - t * V_;
        RA[((size_t)n * CT + (size_t)o * T_ + t) * 32 + v] = f2bf(y);
    }
}

// ---------------- host launcher ----------------
extern "C" void kernel_launch(void* const* d_in, const int* in_sizes, int n_in,
                              void* d_out, int out_size, void* d_ws, size_t ws_size,
                              hipStream_t stream) {
    (void)in_sizes; (void)n_in; (void)out_size; (void)ws_size;
    const float* fv     = (const float*)d_in[0];
    const float* fe     = (const float*)d_in[1];
    const float* srcM   = (const float*)d_in[2];
    const float* tgtM   = (const float*)d_in[3];
    const float* alpha  = (const float*)d_in[4];
    const float* alpha1 = (const float*)d_in[5];
    const float* cen_wc = (const float*)d_in[6];
    const float* cen_bc = (const float*)d_in[7];
    const float* cen_wt = (const float*)d_in[8];
    const float* cen_bt = (const float*)d_in[9];
    const float* bn_g   = (const float*)d_in[10];
    const float* bn_b   = (const float*)d_in[11];
    const float* cen_wv = (const float*)d_in[12];
    const float* cen_bv = (const float*)d_in[13];
    const float* He_w   = (const float*)d_in[14];
    const float* He_b   = (const float*)d_in[15];
    const float* Hv_w   = (const float*)d_in[16];
    const float* Hv_b   = (const float*)d_in[17];
    const float* bne_g  = (const float*)d_in[18];
    const float* bne_b  = (const float*)d_in[19];
    const float* bnv_g  = (const float*)d_in[20];
    const float* bnv_b  = (const float*)d_in[21];

    char* w = (char*)d_ws;
    size_t off = 0;
    auto alloc = [&](size_t bytes) -> void* {
        void* p = w + off;
        off = (off + bytes + 255) & ~(size_t)255;
        return p;
    };
    float* xbuf   = (float*)alloc(800 * 4);
    float* Abuf   = (float*)alloc(20000 * 4);
    float* sMf    = (float*)alloc(640 * 4);
    float* tMf    = (float*)alloc(640 * 4);
    unsigned short* sMB  = (unsigned short*)alloc(1024 * 2);
    unsigned short* tMB  = (unsigned short*)alloc(1024 * 2);
    unsigned short* sMBT = (unsigned short*)alloc(1024 * 2);
    unsigned short* tMBT = (unsigned short*)alloc(1024 * 2);
    unsigned short* HeWb = (unsigned short*)alloc((size_t)C_ * K3C * 2);
    unsigned short* HvWb = (unsigned short*)alloc((size_t)C_ * K3C * 2);
    float* stats   = (float*)alloc(128 * 4);
    float* partial = (float*)alloc((size_t)C_ * N_ * 2 * 4);
    unsigned short* RA    = (unsigned short*)alloc((size_t)N_ * CT * 32 * 2);   // ~39.3 MB
    unsigned short* STACK = (unsigned short*)alloc((size_t)N_ * TVP * 192 * 2); // ~92.2 MB
    float* PRE = (float*)alloc((size_t)N_ * C_ * TV * 4);                       // ~61.4 MB

    float* out_node = (float*)d_out;                       // fvp + fv
    float* out_edge = out_node + (size_t)N_ * C_ * TV;     // fep + fe

    // ---- CeN ----
    cen_reduce<<<N_, 256, 0, stream>>>(fv, fe, cen_wc, cen_bc, cen_wt, cen_bt, xbuf);
    cen_bn_A<<<1, 1024, 0, stream>>>(xbuf, bn_g, bn_b, cen_wv, cen_bv, Abuf);
    cen_finish<<<1, 1024, 0, stream>>>(Abuf, srcM, tgtM, alpha, alpha1, He_w, Hv_w,
                                       sMf, tMf, sMB, tMB, sMBT, tMBT, HeWb, HvWb);

    // ---- edge update ----
    stack_cols<<<dim3(118, N_), 256, 0, stream>>>(fe, STACK);            // rows 0..63 = fe
    to_rowA<<<(N_ * CT) / 8, 256, 0, stream>>>(fv, RA);                  // fv as WMMA A
    graph_wmma<<<dim3(150, N_), 256, 0, stream>>>(RA, sMB, tMB, STACK);  // rows 64..191
    gemm_wmma<<<dim3(235, N_), 256, 0, stream>>>(HeWb, STACK, He_b, PRE);
    bn_stats_partial<<<dim3(C_, N_), 256, 0, stream>>>(PRE, partial);
    bn_stats_combine<<<1, 64, 0, stream>>>(partial, stats);
    bn_finalize<<<60000, 256, 0, stream>>>(PRE, stats, bne_g, bne_b, fe, out_edge, RA, 1);

    // ---- node update (reuses RA=fep, STACK, PRE) ----
    stack_cols<<<dim3(118, N_), 256, 0, stream>>>(fv, STACK);            // rows 0..63 = fv
    graph_wmma<<<dim3(150, N_), 256, 0, stream>>>(RA, sMBT, tMBT, STACK);
    gemm_wmma<<<dim3(235, N_), 256, 0, stream>>>(HvWb, STACK, Hv_b, PRE);
    bn_stats_partial<<<dim3(C_, N_), 256, 0, stream>>>(PRE, partial);
    bn_stats_combine<<<1, 64, 0, stream>>>(partial, stats);
    bn_finalize<<<60000, 256, 0, stream>>>(PRE, stats, bnv_g, bnv_b, fv, out_node,
                                           (unsigned short*)nullptr, 0);
}
